// SPAB_69234872812262
// MI455X (gfx1250) — compile-verified, matrix operand order checked by reading, not run on hardware
//
#include <hip/hip_runtime.h>
#include <hip/hip_bf16.h>
#include <math.h>

typedef __attribute__((ext_vector_type(16))) _Float16 v16h;
typedef __attribute__((ext_vector_type(8)))  _Float16 v8h;
typedef __attribute__((ext_vector_type(8)))  float    v8f;

#define DEV __device__ __forceinline__

static constexpr int CB    = 4;
static constexpr int CDIM  = 192;
static constexpr int CH    = 128, CW = 128;
static constexpr int HWr   = CH * CW;      // 16384 = 2^14
static constexpr int MPIX  = CB * HWr;     // 65536
static constexpr int NHEAD = 6;
static constexpr int HDIM  = 32;
static constexpr int LWIN  = 22;
static constexpr int NWIN  = LWIN * LWIN;  // 484

// ---------- WMMA helpers (CDNA5 wave32, 16x16x32 f16 -> f32) ----------

DEV v16h ldfrag(const _Float16* p) {
  // A/B fragment per CDNA5 layout: elems 0..7 = 8 contiguous K at p,
  // elems 8..15 = 8 contiguous K at p+16 (lane's K-base = (lane>>4)*8).
  v8h lo = *(const v8h*)p;
  v8h hi = *(const v8h*)(p + 16);
  v16h r;
#pragma unroll
  for (int e = 0; e < 8; ++e) { r[e] = lo[e]; r[e + 8] = hi[e]; }
  return r;
}

DEV v8f wmma16(v16h a, v16h b, v8f c) {
  return __builtin_amdgcn_wmma_f32_16x16x32_f16(false, a, false, b,
                                                (short)0, c, false, false);
}

// ---------- generic GEMM: C[M,N] = A[M,K](f16,row) * W[N,K]^T (f16,row) + bias ----------
// MODE 0: out f16 pixel-major [M,N]
// MODE 1: same + exact GELU
// MODE 2: out f32 channel-major (B,N,H,W)
// MODE 3: MODE 2 + residual add (final output)
template <int MODE>
__global__ __launch_bounds__(256) void gemm_k(
    const _Float16* __restrict__ A, const _Float16* __restrict__ W,
    const float* __restrict__ bias, int K, int N,
    _Float16* __restrict__ outH, float* __restrict__ outF,
    const float* __restrict__ resid) {
  const int lane  = threadIdx.x & 31;
  const int wid   = threadIdx.x >> 5;          // 8 waves: 4 (M) x 2 (N)
  const int Morg  = blockIdx.x * 128 + (wid >> 1) * 32;
  const int Norg  = blockIdx.y * 64 + (wid & 1) * 32;
  const int mrow  = lane & 15;
  const int khalf = lane >> 4;

  v8f acc00 = {}, acc01 = {}, acc10 = {}, acc11 = {};
  const _Float16* ar0 = A + (size_t)(Morg + mrow) * K + khalf * 8;
  const _Float16* ar1 = A + (size_t)(Morg + 16 + mrow) * K + khalf * 8;
  const _Float16* br0 = W + (size_t)(Norg + mrow) * K + khalf * 8;
  const _Float16* br1 = W + (size_t)(Norg + 16 + mrow) * K + khalf * 8;

  for (int k0 = 0; k0 < K; k0 += 32) {
    v16h a0 = ldfrag(ar0 + k0);
    v16h a1 = ldfrag(ar1 + k0);
    v16h b0 = ldfrag(br0 + k0);
    v16h b1 = ldfrag(br1 + k0);
    acc00 = wmma16(a0, b0, acc00);
    acc01 = wmma16(a0, b1, acc01);
    acc10 = wmma16(a1, b0, acc10);
    acc11 = wmma16(a1, b1, acc11);
  }

#pragma unroll
  for (int mi = 0; mi < 2; ++mi) {
#pragma unroll
    for (int ni = 0; ni < 2; ++ni) {
      const v8f& ac = (mi == 0) ? (ni == 0 ? acc00 : acc01)
                                : (ni == 0 ? acc10 : acc11);
      const int n  = Norg + ni * 16 + mrow;
      const float bi = bias[n];
#pragma unroll
      for (int r = 0; r < 8; ++r) {
        const int m = Morg + mi * 16 + khalf * 8 + r;
        float v = ac[r] + bi;
        if (MODE == 1) v = 0.5f * v * (1.0f + erff(v * 0.70710678118654752f));
        if (MODE == 0 || MODE == 1) {
          outH[(size_t)m * N + n] = (_Float16)v;
        } else {
          const size_t idx =
              ((size_t)((m >> 14) * N + n) << 14) + (size_t)(m & (HWr - 1));
          if (MODE == 3) v += resid[idx];
          outF[idx] = v;
        }
      }
    }
  }
}

// ---------- LayerNorm over channels: f32 channel-major -> f16 pixel-major ----------
__global__ void ln_cm_k(const float* __restrict__ X, const float* __restrict__ g,
                        const float* __restrict__ bt, _Float16* __restrict__ out,
                        int C) {
  const int m = blockIdx.x * blockDim.x + threadIdx.x;
  if (m >= MPIX) return;
  const int b = m >> 14, pix = m & (HWr - 1);
  const float* base = X + ((size_t)b * C << 14) + pix;
  float s = 0.f, ss = 0.f;
  for (int c = 0; c < C; ++c) {
    float v = base[(size_t)c << 14];
    s += v; ss += v * v;
  }
  const float mu = s / (float)C;
  const float var = ss / (float)C - mu * mu;
  const float rs = rsqrtf(var + 1e-5f);
  _Float16* o = out + (size_t)m * C;
  for (int c = 0; c < C; ++c) {
    float v = base[(size_t)c << 14];
    o[c] = (_Float16)((v - mu) * rs * g[c] + bt[c]);
  }
}

// ---------- LayerNorm of z2 (cols 384..767 of Z, f16 PM) -> f16 channel-major ----------
__global__ void ln_z2_k(const _Float16* __restrict__ Z, const float* __restrict__ g,
                        const float* __restrict__ bt, _Float16* __restrict__ out) {
  const int m = blockIdx.x * blockDim.x + threadIdx.x;
  if (m >= MPIX) return;
  const int b = m >> 14, pix = m & (HWr - 1);
  const _Float16* row = Z + (size_t)m * 768 + 384;
  float s = 0.f, ss = 0.f;
  for (int c = 0; c < 384; ++c) {
    float v = (float)row[c];
    s += v; ss += v * v;
  }
  const float mu = s * (1.0f / 384.0f);
  const float var = ss * (1.0f / 384.0f) - mu * mu;
  const float rs = rsqrtf(var + 1e-5f);
  for (int c = 0; c < 384; ++c) {
    float v = ((float)row[c] - mu) * rs * g[c] + bt[c];
    out[((size_t)(b * 384 + c) << 14) + pix] = (_Float16)v;
  }
}

// ---------- attention: one wave per (window, head) ----------
__global__ __launch_bounds__(32) void attn_k(
    const _Float16* __restrict__ Y, const float* __restrict__ temp,
    const float* __restrict__ rpb, float* __restrict__ o_acc,
    float* __restrict__ s_acc) {
  __shared__ __align__(16) _Float16 qs[64][32];
  __shared__ __align__(16) _Float16 ks[64][32];
  __shared__ __align__(16) _Float16 vt[32][64];  // v transposed [c][pixel]
  __shared__ __align__(16) float    at[64][64];
  __shared__ __align__(16) _Float16 el[64][64];
  __shared__ float rb[1350];

  const int lane = threadIdx.x;
  const int h    = blockIdx.x % NHEAD;
  const int widx = blockIdx.x / NHEAD;
  const int b    = widx / NWIN;
  const int l    = widx % NWIN;
  const int lh   = l / LWIN, lw = l % LWIN;

  for (int i = lane; i < 1350; i += 32) rb[i] = rpb[i];
  const float tf = temp[h];

#pragma unroll
  for (int pp = 0; pp < 2; ++pp) {
    const int p  = lane + pp * 32;
    const int iy = p >> 3, ix = p & 7;
    const int gh = lh * 6 - 3 + iy, gw = lw * 6 - 3 + ix;
    if (gh >= 0 && gh < CH && gw >= 0 && gw < CW) {
      const _Float16* base = Y + (size_t)((b << 14) + gh * CW + gw) * 576;
#pragma unroll
      for (int e = 0; e < 32; e += 8) {
        v8h vq = *(const v8h*)(base + h * 32 + e);
#pragma unroll
        for (int t = 0; t < 8; ++t)
          qs[p][e + t] = (_Float16)((float)vq[t] * tf);
        v8h vk = *(const v8h*)(base + 192 + h * 32 + e);
        *(v8h*)&ks[p][e] = vk;
      }
#pragma unroll
      for (int e = 0; e < 32; ++e) vt[e][p] = base[384 + h * 32 + e];
    } else {
      for (int e = 0; e < 32; ++e) {
        qs[p][e] = (_Float16)0.f;
        ks[p][e] = (_Float16)0.f;
        vt[e][p] = (_Float16)0.f;
      }
    }
  }
  __syncthreads();

  const int mrow = lane & 15, khalf = lane >> 4;

  // attn = q (64x32) * k^T (32x64): 16 WMMAs
  for (int mi = 0; mi < 4; ++mi) {
    v16h af = ldfrag(&qs[mi * 16 + mrow][khalf * 8]);
    for (int ni = 0; ni < 4; ++ni) {
      v16h bf = ldfrag(&ks[ni * 16 + mrow][khalf * 8]);
      v8f ac = {};
      ac = wmma16(af, bf, ac);
#pragma unroll
      for (int r = 0; r < 8; ++r)
        at[mi * 16 + khalf * 8 + r][ni * 16 + mrow] = ac[r];
    }
  }
  __syncthreads();

  // bias + exp + row sums; scatter s into s_acc (fold)
#pragma unroll
  for (int rr = 0; rr < 2; ++rr) {
    const int i  = lane + rr * 32;
    const int iy = i >> 3, ix = i & 7;
    float mx = -3.0e38f;
    for (int j = 0; j < 64; ++j) {
      const int jy = j >> 3, jx = j & 7;
      float v = at[i][j] + rb[((iy - jy + 7) * 15 + (ix - jx + 7)) * 6 + h];
      at[i][j] = v;
      mx = fmaxf(mx, v);
    }
    float s = 0.f;
    for (int j = 0; j < 64; ++j) {
      float e = expf(at[i][j] - mx);
      s += e;
      el[i][j] = (_Float16)e;
    }
    const int gh = lh * 6 - 3 + iy, gw = lw * 6 - 3 + ix;
    if (gh >= 0 && gh < CH && gw >= 0 && gw < CW)
      atomicAdd(&s_acc[((size_t)(b * NHEAD + h) << 14) + gh * CW + gw], s);
  }
  __syncthreads();

  // o = e (64x64) * v (64x32): 16 WMMAs; scatter into o_acc (fold)
  for (int mi = 0; mi < 4; ++mi) {
    v8f ac0 = {}, ac1 = {};
#pragma unroll
    for (int kk = 0; kk < 2; ++kk) {
      v16h af = ldfrag(&el[mi * 16 + mrow][kk * 32 + khalf * 8]);
      v16h b0 = ldfrag(&vt[mrow][kk * 32 + khalf * 8]);
      v16h b1 = ldfrag(&vt[16 + mrow][kk * 32 + khalf * 8]);
      ac0 = wmma16(af, b0, ac0);
      ac1 = wmma16(af, b1, ac1);
    }
#pragma unroll
    for (int r = 0; r < 8; ++r) {
      const int i  = mi * 16 + khalf * 8 + r;
      const int iy = i >> 3, ix = i & 7;
      const int gh = lh * 6 - 3 + iy, gw = lw * 6 - 3 + ix;
      if (gh >= 0 && gh < CH && gw >= 0 && gw < CW) {
        const size_t pix = (size_t)gh * CW + gw;
        atomicAdd(&o_acc[((size_t)(b * CDIM + h * HDIM + mrow) << 14) + pix],
                  ac0[r]);
        atomicAdd(
            &o_acc[((size_t)(b * CDIM + h * HDIM + 16 + mrow) << 14) + pix],
            ac1[r]);
      }
    }
  }
}

// ---------- out = o_acc / s_acc (per head) -> f16 pixel-major ----------
__global__ void outn_k(const float* __restrict__ o_acc,
                       const float* __restrict__ s_acc,
                       _Float16* __restrict__ out) {
  const int m = blockIdx.x * blockDim.x + threadIdx.x;
  if (m >= MPIX) return;
  const int b = m >> 14, pix = m & (HWr - 1);
  float sv[NHEAD];
#pragma unroll
  for (int h = 0; h < NHEAD; ++h)
    sv[h] = s_acc[((size_t)(b * NHEAD + h) << 14) + pix];
  _Float16* o = out + (size_t)m * CDIM;
  for (int c = 0; c < CDIM; ++c)
    o[c] = (_Float16)(o_acc[((size_t)(b * CDIM + c) << 14) + pix] / sv[c >> 5]);
}

// ---------- hft pooling: a = 3p - up(pool2) - up(pool4) - up(pool8) ----------
__global__ void hft_pool_k(const float* __restrict__ P, float* __restrict__ A) {
  const int gid = blockIdx.x * blockDim.x + threadIdx.x;  // one 8x8 block each
  if (gid >= CB * CDIM * 256) return;
  const int bc = gid >> 8;
  const int blk = gid & 255;
  const int by = blk >> 4, bx = blk & 15;
  const float* base = P + ((size_t)bc << 14) + (size_t)(by * 8) * CW + bx * 8;
  float v[8][8];
  float s2[4][4] = {}, s4[2][2] = {}, s8 = 0.f;
  for (int y = 0; y < 8; ++y)
    for (int x = 0; x < 8; ++x) {
      float t = base[y * CW + x];
      v[y][x] = t;
      s2[y >> 1][x >> 1] += t;
    }
  for (int y = 0; y < 4; ++y)
    for (int x = 0; x < 4; ++x) s4[y >> 1][x >> 1] += s2[y][x];
  for (int y = 0; y < 2; ++y)
    for (int x = 0; x < 2; ++x) s8 += s4[y][x];
  float* ob = A + ((size_t)bc << 14) + (size_t)(by * 8) * CW + bx * 8;
  for (int y = 0; y < 8; ++y)
    for (int x = 0; x < 8; ++x)
      ob[y * CW + x] = 3.f * v[y][x] - s2[y >> 1][x >> 1] * 0.25f -
                       s4[y >> 2][x >> 2] * 0.0625f - s8 * 0.015625f;
}

// ---------- hft gate: x1 = x + sigmoid(dwconv3(a)+b) * p ----------
__global__ void hft_gate_k(const float* __restrict__ A, const float* __restrict__ P,
                           const float* __restrict__ X, const float* __restrict__ hw,
                           const float* __restrict__ hb, float* __restrict__ X1) {
  const size_t idx = (size_t)blockIdx.x * blockDim.x + threadIdx.x;
  if (idx >= (size_t)CB * CDIM * HWr) return;
  const int pix = (int)(idx & (HWr - 1));
  const int bc = (int)(idx >> 14);
  const int c = bc % CDIM;
  const int ph = pix >> 7, pw = pix & 127;
  const float* base = A + ((size_t)bc << 14);
  float acc = hb[c];
#pragma unroll
  for (int ky = 0; ky < 3; ++ky)
#pragma unroll
    for (int kx = 0; kx < 3; ++kx) {
      const int yy = ph + ky - 1, xx = pw + kx - 1;
      if (yy >= 0 && yy < CH && xx >= 0 && xx < CW)
        acc += hw[c * 9 + ky * 3 + kx] * base[yy * CW + xx];
    }
  const float sg = 1.f / (1.f + expf(-acc));
  X1[idx] = X[idx] + sg * P[idx];
}

// ---------- z1 * (dwconv3(z2n)+b) -> f16 pixel-major A2[M,384] ----------
__global__ void dwmul_k(const _Float16* __restrict__ z2n,
                        const _Float16* __restrict__ Z,
                        const float* __restrict__ sw, const float* __restrict__ sb,
                        _Float16* __restrict__ A2) {
  const size_t idx = (size_t)blockIdx.x * blockDim.x + threadIdx.x;
  if (idx >= (size_t)CB * 384 * HWr) return;
  const int pix = (int)(idx & (HWr - 1));
  const int bc = (int)(idx >> 14);
  const int c = bc % 384;
  const int b = bc / 384;
  const int ph = pix >> 7, pw = pix & 127;
  const _Float16* base = z2n + ((size_t)bc << 14);
  float acc = sb[c];
#pragma unroll
  for (int ky = 0; ky < 3; ++ky)
#pragma unroll
    for (int kx = 0; kx < 3; ++kx) {
      const int yy = ph + ky - 1, xx = pw + kx - 1;
      if (yy >= 0 && yy < CH && xx >= 0 && xx < CW)
        acc += sw[c * 9 + ky * 3 + kx] * (float)base[yy * CW + xx];
    }
  const int m = (b << 14) + pix;
  const float z1 = (float)Z[(size_t)m * 768 + c];
  A2[(size_t)m * 384 + c] = (_Float16)(z1 * acc);
}

// ---------- f32 -> f16 weight conversion ----------
__global__ void cvt_k(const float* __restrict__ s, _Float16* __restrict__ d,
                      int n) {
  const int i = blockIdx.x * blockDim.x + threadIdx.x;
  if (i < n) d[i] = (_Float16)s[i];
}

// =====================================================================

extern "C" void kernel_launch(void* const* d_in, const int* in_sizes, int n_in,
                              void* d_out, int out_size, void* d_ws,
                              size_t ws_size, hipStream_t stream) {
  const float* x      = (const float*)d_in[0];
  const float* n1g    = (const float*)d_in[1];
  const float* n1b    = (const float*)d_in[2];
  const float* qkv_w  = (const float*)d_in[3];
  const float* qkv_b  = (const float*)d_in[4];
  const float* temp   = (const float*)d_in[5];
  const float* rpb    = (const float*)d_in[6];
  const float* proj_w = (const float*)d_in[7];
  const float* proj_b = (const float*)d_in[8];
  const float* hft_w  = (const float*)d_in[9];
  const float* hft_b  = (const float*)d_in[10];
  const float* n2g    = (const float*)d_in[11];
  const float* n2b    = (const float*)d_in[12];
  const float* fc1_w  = (const float*)d_in[13];
  const float* fc1_b  = (const float*)d_in[14];
  const float* sgn_g  = (const float*)d_in[15];
  const float* sgn_b  = (const float*)d_in[16];
  const float* sg_w   = (const float*)d_in[17];
  const float* sg_b   = (const float*)d_in[18];
  const float* fc2_w  = (const float*)d_in[19];
  const float* fc2_b  = (const float*)d_in[20];

  char* ws = (char*)d_ws;
  // Region plan (bytes), reused across phases:
  constexpr size_t SZ_RA = (size_t)MPIX * 384 * 2;  // 50,331,648
  constexpr size_t SZ_RY = (size_t)MPIX * 576 * 2;  // 75,497,472
  constexpr size_t SZ_RX = (size_t)MPIX * 192 * 4;  // 50,331,648
  constexpr size_t SZ_RZ = (size_t)MPIX * 768 * 2;  // 100,663,296
  constexpr size_t SZ_RO = (size_t)MPIX * 192 * 4;  // 50,331,648
  constexpr size_t SZ_RS = (size_t)MPIX * 6 * 4;    // 1,572,864
  constexpr size_t OFF_RA = 0;
  constexpr size_t OFF_RY = OFF_RA + SZ_RA;
  constexpr size_t OFF_RX = OFF_RY + SZ_RY;
  constexpr size_t OFF_RZ = OFF_RX + SZ_RX;
  constexpr size_t OFF_RO = OFF_RZ + SZ_RZ;
  constexpr size_t OFF_RS = OFF_RO + SZ_RO;
  constexpr size_t OFF_WQ = OFF_RS + SZ_RS;
  constexpr size_t OFF_WP = OFF_WQ + (size_t)576 * 192 * 2;
  constexpr size_t OFF_WF1 = OFF_WP + (size_t)192 * 192 * 2;
  constexpr size_t OFF_WF2 = OFF_WF1 + (size_t)768 * 192 * 2;

  _Float16* xn   = (_Float16*)(ws + OFF_RA);  // [M,192] f16 PM
  _Float16* Y    = (_Float16*)(ws + OFF_RY);  // [M,576] f16 PM
  float*    Pbuf = (float*)(ws + OFF_RY);     // proj out, f32 CM (reuses RY)
  _Float16* z2n  = (_Float16*)(ws + OFF_RY);  // f16 CM 384ch (reuses RY)
  float*    Abuf = (float*)(ws + OFF_RA);     // hft 'a', f32 CM (reuses RA)
  _Float16* zin  = (_Float16*)(ws + OFF_RA);  // ln2 out f16 PM (reuses RA)
  _Float16* A2   = (_Float16*)(ws + OFF_RA);  // [M,384] f16 PM (reuses RA)
  float*    x1   = (float*)(ws + OFF_RX);
  _Float16* Zb   = (_Float16*)(ws + OFF_RZ);  // gelu(fc1) [M,768] f16 PM
  float*    oacc = (float*)(ws + OFF_RO);
  float*    sacc = (float*)(ws + OFF_RS);
  _Float16* wq   = (_Float16*)(ws + OFF_WQ);
  _Float16* wp   = (_Float16*)(ws + OFF_WP);
  _Float16* wf1  = (_Float16*)(ws + OFF_WF1);
  _Float16* wf2  = (_Float16*)(ws + OFF_WF2);

  const dim3 B256(256);

  // weight conversions (f32 -> f16)
  cvt_k<<<(576 * 192 + 255) / 256, B256, 0, stream>>>(qkv_w, wq, 576 * 192);
  cvt_k<<<(192 * 192 + 255) / 256, B256, 0, stream>>>(proj_w, wp, 192 * 192);
  cvt_k<<<(768 * 192 + 255) / 256, B256, 0, stream>>>(fc1_w, wf1, 768 * 192);
  cvt_k<<<(192 * 384 + 255) / 256, B256, 0, stream>>>(fc2_w, wf2, 192 * 384);

  // LN1 -> xn
  ln_cm_k<<<MPIX / 256, B256, 0, stream>>>(x, n1g, n1b, xn, 192);

  // qkv GEMM: Y[M,576] = xn * Wqkv^T + b
  gemm_k<0><<<dim3(MPIX / 128, 576 / 64), B256, 0, stream>>>(
      xn, wq, qkv_b, 192, 576, Y, nullptr, nullptr);

  // zero fold accumulators
  hipMemsetAsync(oacc, 0, SZ_RO, stream);
  hipMemsetAsync(sacc, 0, SZ_RS, stream);

  // windowed attention + overlap-add fold
  attn_k<<<CB * NWIN * NHEAD, dim3(32), 0, stream>>>(Y, temp, rpb, oacc, sacc);

  // normalize fold -> f16 PM
  outn_k<<<MPIX / 256, B256, 0, stream>>>(oacc, sacc, xn /* reuse RA */);

  // proj GEMM -> P f32 CM
  gemm_k<2><<<dim3(MPIX / 128, 192 / 64), B256, 0, stream>>>(
      xn, wp, proj_b, 192, 192, nullptr, Pbuf, nullptr);

  // hft: a = 3p - pools (RA now free)
  hft_pool_k<<<(CB * CDIM * 256) / 256, B256, 0, stream>>>(Pbuf, Abuf);
  // x1 = x + sigmoid(dw(a)) * p
  hft_gate_k<<<(CB * CDIM * HWr) / 256, B256, 0, stream>>>(Abuf, Pbuf, x, hft_w,
                                                           hft_b, x1);

  // LN2 -> zin f16 PM (RA)
  ln_cm_k<<<MPIX / 256, B256, 0, stream>>>(x1, n2g, n2b, zin, 192);

  // fc1 GEMM + exact GELU -> Z[M,768] f16 PM
  gemm_k<1><<<dim3(MPIX / 128, 768 / 64), B256, 0, stream>>>(
      zin, wf1, fc1_b, 192, 768, Zb, nullptr, nullptr);

  // LN over z2 channels -> z2n f16 CM (RY now free)
  ln_z2_k<<<MPIX / 256, B256, 0, stream>>>(Zb, sgn_g, sgn_b, z2n);

  // z1 * (dwconv3(z2n)+b) -> A2[M,384] f16 PM (RA free)
  dwmul_k<<<(CB * 384 * HWr) / 256, B256, 0, stream>>>(z2n, Zb, sg_w, sg_b, A2);

  // fc2 GEMM + residual x1 -> d_out f32 channel-major
  gemm_k<3><<<dim3(MPIX / 128, 192 / 64), B256, 0, stream>>>(
      A2, wf2, fc2_b, 384, 192, nullptr, (float*)d_out, x1);

  (void)in_sizes; (void)n_in; (void)out_size; (void)ws_size;
}
